// ASAGNN_23381801959633
// MI455X (gfx1250) — matrix operand order repaired and checked by reading, not compiled, not run on hardware
//
#include <hip/hip_runtime.h>
#include <hip/hip_bf16.h>
#include <stdint.h>

// Problem constants (match reference)
#define NN 8192   // nodes
#define DD 128    // features
#define NW 256    // NN/32 mask words per row

// Flip to 0 if the async-to-LDS mnemonic is rejected by the assembler.
#define USE_ASYNC_LDS 1

typedef __attribute__((ext_vector_type(16))) __bf16    v16bf;
typedef __attribute__((ext_vector_type(8)))  float     v8f;
typedef __attribute__((ext_vector_type(4)))  uint32_t  u32x4;
typedef __attribute__((ext_vector_type(2)))  uint32_t  u32x2;
typedef __attribute__((ext_vector_type(4)))  float     f32x4;

union Frag { uint32_t u[8]; u32x4 q[2]; v16bf v; };

__device__ __forceinline__ uint32_t f2bf(float f) {           // f32 -> bf16 bits (RNE)
  uint32_t u = __float_as_uint(f);
  return (u + 0x7FFFu + ((u >> 16) & 1u)) >> 16;
}
__device__ __forceinline__ uint32_t pk2(float lo, float hi) { return f2bf(lo) | (f2bf(hi) << 16); }
__device__ __forceinline__ float    bf2f(uint32_t h) { return __uint_as_float(h << 16); }

__device__ __forceinline__ void async_copy16(const uint16_t* g, uint16_t* lds_generic) {
#if USE_ASYNC_LDS
  // LDS address comes from the first VGPR operand; low 32 bits of the generic shared pointer
  // are the LDS byte offset (aperture rule, ISA 10.2).
  uint32_t loff = (uint32_t)(uintptr_t)lds_generic;
  uint64_t ga   = (uint64_t)(uintptr_t)g;
  asm volatile("global_load_async_to_lds_b128 %0, %1, off" :: "v"(loff), "v"(ga) : "memory");
#else
  *(u32x4*)lds_generic = *(const u32x4*)g;
#endif
}
__device__ __forceinline__ void async_wait0() {
#if USE_ASYNC_LDS
  asm volatile("s_wait_asynccnt 0x0" ::: "memory");
#endif
}

// ---------------------------------------------------------------------------
// Pack: xn = bf16(normalized x) row-major; htx = bf16(x) transposed [D][N].
__global__ void pack_x_kernel(const float* __restrict__ x,
                              uint16_t* __restrict__ xn,
                              uint16_t* __restrict__ htx) {
  int wid  = (blockIdx.x * blockDim.x + threadIdx.x) >> 5;
  int lane = threadIdx.x & 31;
  f32x4 xv = *(const f32x4*)(x + (size_t)wid * DD + lane * 4);
  float ss = xv[0]*xv[0] + xv[1]*xv[1] + xv[2]*xv[2] + xv[3]*xv[3];
#pragma unroll
  for (int m = 16; m >= 1; m >>= 1) ss += __shfl_xor(ss, m, 32);
  float inv = rsqrtf(ss + 1e-12f);
  u32x2 xnw; xnw[0] = pk2(xv[0]*inv, xv[1]*inv); xnw[1] = pk2(xv[2]*inv, xv[3]*inv);
  *(u32x2*)(xn + (size_t)wid * DD + lane * 4) = xnw;
  htx[(size_t)(lane*4+0) * NN + wid] = (uint16_t)f2bf(xv[0]);
  htx[(size_t)(lane*4+1) * NN + wid] = (uint16_t)f2bf(xv[1]);
  htx[(size_t)(lane*4+2) * NN + wid] = (uint16_t)f2bf(xv[2]);
  htx[(size_t)(lane*4+3) * NN + wid] = (uint16_t)f2bf(xv[3]);
}

// Wt[n][k] = bf16(W[k][n])  (B-operand wants columns of W contiguous)
__global__ void pack_w_kernel(const float* __restrict__ W, uint16_t* __restrict__ Wt) {
  int idx = blockIdx.x * blockDim.x + threadIdx.x;   // 16384
  int n = idx >> 7, k = idx & 127;
  Wt[idx] = (uint16_t)f2bf(W[(size_t)k * DD + n]);
}

// adj (int32, 256MB, read exactly once) -> bitmask (8MB).
__global__ void adj_pack_kernel(const int* __restrict__ adj, uint32_t* __restrict__ adjbits) {
  int wid  = (blockIdx.x * blockDim.x + threadIdx.x) >> 5;
  int lane = threadIdx.x & 31;
  int i = wid >> 8, w = wid & 255;
  int a = adj[(size_t)i * NN + w * 32 + lane];
  uint32_t m = __builtin_amdgcn_ballot_w32(a > 0);
  if (lane == 0) adjbits[(size_t)i * NW + w] = m;
}

// ---------------------------------------------------------------------------
// sim = xn@xn^T via WMMA bf16; maskbits = (sim > 0.5) & adjbits.
__global__ __launch_bounds__(256)
void sim_kernel(const uint16_t* __restrict__ xn,
                const uint32_t* __restrict__ adjbits,
                uint32_t* __restrict__ maskbits) {
  int gw   = (blockIdx.x * blockDim.x + threadIdx.x) >> 5;
  int lane = threadIdx.x & 31;
  int it = gw >> 8;          // 512 i-tiles
  int js = gw & 255;         // 256 j-strips (32 cols)
  int i0 = it * 16, j0 = js * 32;
  int lr = lane & 15, lh = lane >> 4;
  v8f c0, c1;
#pragma unroll
  for (int e = 0; e < 8; ++e) { c0[e] = 0.f; c1[e] = 0.f; }
  const uint16_t* ar  = xn + (size_t)(i0 + lr) * DD + lh * 16;
  const uint16_t* br0 = xn + (size_t)(j0 + lr) * DD + lh * 16;
  const uint16_t* br1 = xn + (size_t)(j0 + 16 + lr) * DD + lh * 16;
#pragma unroll
  for (int k0 = 0; k0 < DD; k0 += 32) {
    Frag a, b0, b1;
    a.q[0]  = *(const u32x4*)(ar + k0);
    a.q[1]  = *(const u32x4*)(ar + k0 + 8);
    b0.q[0] = *(const u32x4*)(br0 + k0);
    b0.q[1] = *(const u32x4*)(br0 + k0 + 8);
    b1.q[0] = *(const u32x4*)(br1 + k0);
    b1.q[1] = *(const u32x4*)(br1 + k0 + 8);
    c0 = __builtin_amdgcn_wmma_f32_16x16x32_bf16(false, a.v, false, b0.v, (short)0, c0, false, false);
    c1 = __builtin_amdgcn_wmma_f32_16x16x32_bf16(false, a.v, false, b1.v, (short)0, c1, false, false);
  }
  uint32_t wlo[8], whi[8];
#pragma unroll
  for (int r = 0; r < 8; ++r) {
    uint32_t m0 = __builtin_amdgcn_ballot_w32(c0[r] > 0.5f);
    uint32_t m1 = __builtin_amdgcn_ballot_w32(c1[r] > 0.5f);
    wlo[r] = (m0 & 0xFFFFu) | (m1 << 16);
    whi[r] = (m0 >> 16)     | (m1 & 0xFFFF0000u);
  }
  if (lane == 0) {
#pragma unroll
    for (int r = 0; r < 8; ++r) {
      size_t o0 = (size_t)(i0 + r) * NW + js;
      size_t o1 = (size_t)(i0 + 8 + r) * NW + js;
      maskbits[o0] = wlo[r] & adjbits[o0];
      maskbits[o1] = whi[r] & adjbits[o1];
    }
  }
}

// invdeg[i] = 1 / max(popcount(mask row i), 1)
__global__ void deg_kernel(const uint32_t* __restrict__ maskbits, float* __restrict__ invdeg) {
  int i = blockIdx.x * blockDim.x + threadIdx.x;
  const uint32_t* row = maskbits + (size_t)i * NW;
  int s = 0;
#pragma unroll 8
  for (int w = 0; w < NW; ++w) s += __popc(row[w]);
  invdeg[i] = 1.0f / fmaxf((float)s, 1.0f);
}

// ---------------------------------------------------------------------------
// s = h + (maskf @ h) * invdeg.
// Block = 8 waves / 2 i-tiles; wave handles (i-tile, 2 N-tiles).
// ht K-slices (128x32 bf16 = 8KB) staged in LDS, double-buffered with
// async global->LDS copies overlapped against WMMA on the current buffer.
__global__ __launch_bounds__(256)
void agg_kernel(const uint32_t* __restrict__ maskbits,
                const uint16_t* __restrict__ ht,      // bf16 h transposed [D][N]
                const float* __restrict__ invdeg,
                uint16_t* __restrict__ sout) {        // bf16 row-major [N][D]
  __shared__ uint16_t lbuf[2][DD * 32];               // 2 x 8KB
  int tid  = threadIdx.x;
  int w    = tid >> 5, lane = tid & 31;
  int lr   = lane & 15, lh = lane >> 4;
  int i0   = (blockIdx.x * 2 + (w >> 2)) * 16;        // 2 i-tiles per block
  int t0   = (w & 3) * 2;                              // N-tiles t0, t0+1
  v8f acc0, acc1;
#pragma unroll
  for (int e = 0; e < 8; ++e) { acc0[e] = 0.f; acc1[e] = 0.f; }

  const uint32_t* mrow = maskbits + (size_t)(i0 + lr) * NW;
  // Staging: thread copies 32B of one feature row (2 threads per 64B row).
  int sn = tid >> 1, spart = (tid & 1) * 16;
  const uint16_t* gstage = ht + (size_t)sn * NN + spart;
  uint16_t* lstage0 = &lbuf[0][sn * 32 + spart];
  uint16_t* lstage1 = &lbuf[1][sn * 32 + spart];

  // Preload k-step 0 into buffer 0.
  async_copy16(gstage, lstage0);
  async_copy16(gstage + 8, lstage0 + 8);
  async_wait0();
  __syncthreads();

  for (int kt = 0; kt < NW; ++kt) {
    int cur = kt & 1;
    // Stage next slice into the other buffer (overlapped with compute).
    if (kt + 1 < NW) {
      const uint16_t* g = gstage + (kt + 1) * 32;
      uint16_t* l = cur ? lstage0 : lstage1;
      async_copy16(g, l);
      async_copy16(g + 8, l + 8);
    }
    // Synthesize A-fragment (bf16 1.0/0.0) from mask bits.
    uint32_t wd  = mrow[kt];
    uint32_t byA = (wd >> (16 * lh)) & 0xFFu;
    uint32_t byB = (wd >> (16 * lh + 8)) & 0xFFu;
    Frag a;
#pragma unroll
    for (int v = 0; v < 4; ++v) {
      a.u[v]     = (((byA >> (2*v  )) & 1u) ? 0x00003F80u : 0u) |
                   (((byA >> (2*v+1)) & 1u) ? 0x3F800000u : 0u);
      a.u[4 + v] = (((byB >> (2*v  )) & 1u) ? 0x00003F80u : 0u) |
                   (((byB >> (2*v+1)) & 1u) ? 0x3F800000u : 0u);
    }
    // B-fragments from LDS (ds_load_b128 x2 each).
    const uint16_t* l0 = &lbuf[cur][(t0 * 16 + lr) * 32 + lh * 16];
    const uint16_t* l1 = &lbuf[cur][((t0 + 1) * 16 + lr) * 32 + lh * 16];
    Frag b0, b1;
    b0.q[0] = *(const u32x4*)(l0);  b0.q[1] = *(const u32x4*)(l0 + 8);
    b1.q[0] = *(const u32x4*)(l1);  b1.q[1] = *(const u32x4*)(l1 + 8);
    acc0 = __builtin_amdgcn_wmma_f32_16x16x32_bf16(false, a.v, false, b0.v, (short)0, acc0, false, false);
    acc1 = __builtin_amdgcn_wmma_f32_16x16x32_bf16(false, a.v, false, b1.v, (short)0, acc1, false, false);
    // Async copies for next buffer must land before anyone reads it; and all
    // reads of `cur` are done (loads consumed by WMMA) before it is reused.
    async_wait0();
    __syncthreads();
  }

  float invd[8];
  {
    f32x4 v0 = *(const f32x4*)(invdeg + i0 + 8 * lh);
    f32x4 v1 = *(const f32x4*)(invdeg + i0 + 8 * lh + 4);
#pragma unroll
    for (int e = 0; e < 4; ++e) { invd[e] = v0[e]; invd[4 + e] = v1[e]; }
  }
#pragma unroll
  for (int t = 0; t < 2; ++t) {
    int nt = t0 + t;
    u32x4 hq = *(const u32x4*)(ht + (size_t)(nt * 16 + lr) * NN + i0 + 8 * lh);
    v8f acc = t ? acc1 : acc0;
#pragma unroll
    for (int e = 0; e < 8; ++e) {
      float hval = bf2f((hq[e >> 1] >> ((e & 1) * 16)) & 0xFFFFu);
      float sv   = hval + acc[e] * invd[e];
      sout[(size_t)(i0 + e + 8 * lh) * DD + nt * 16 + lr] = (uint16_t)f2bf(sv);
    }
  }
}

// ---------------------------------------------------------------------------
// hnew = relu(s @ W + b); emit row-major (hb) and transposed (ht) bf16.
// Wave handles (i-tile, 2 N-tiles): 2048 waves.
__global__ __launch_bounds__(256)
void upd_kernel(const uint16_t* __restrict__ sin,   // bf16 [N][D]
                const uint16_t* __restrict__ Wt,    // bf16 [D][D] = W^T
                const float* __restrict__ bias,
                uint16_t* __restrict__ hb_out,      // bf16 [N][D]
                uint16_t* __restrict__ ht_out) {    // bf16 [D][N]
  int gw   = (blockIdx.x * blockDim.x + threadIdx.x) >> 5;  // 0..2047
  int lane = threadIdx.x & 31;
  int lr = lane & 15, lh = lane >> 4;
  int i0 = (gw >> 2) * 16;
  int t0 = (gw & 3) * 2;
  v8f acc0, acc1;
#pragma unroll
  for (int e = 0; e < 8; ++e) { acc0[e] = 0.f; acc1[e] = 0.f; }
  const uint16_t* ar  = sin + (size_t)(i0 + lr) * DD + lh * 16;
  const uint16_t* br0 = Wt + (size_t)(t0 * 16 + lr) * DD + lh * 16;
  const uint16_t* br1 = Wt + (size_t)((t0 + 1) * 16 + lr) * DD + lh * 16;
#pragma unroll
  for (int k0 = 0; k0 < DD; k0 += 32) {
    Frag a, b0, b1;
    a.q[0]  = *(const u32x4*)(ar + k0);
    a.q[1]  = *(const u32x4*)(ar + k0 + 8);
    b0.q[0] = *(const u32x4*)(br0 + k0); b0.q[1] = *(const u32x4*)(br0 + k0 + 8);
    b1.q[0] = *(const u32x4*)(br1 + k0); b1.q[1] = *(const u32x4*)(br1 + k0 + 8);
    acc0 = __builtin_amdgcn_wmma_f32_16x16x32_bf16(false, a.v, false, b0.v, (short)0, acc0, false, false);
    acc1 = __builtin_amdgcn_wmma_f32_16x16x32_bf16(false, a.v, false, b1.v, (short)0, acc1, false, false);
  }
#pragma unroll
  for (int t = 0; t < 2; ++t) {
    int nt = t0 + t;
    v8f acc = t ? acc1 : acc0;
    float bv = bias[nt * 16 + lr];
    float hv[8];
#pragma unroll
    for (int e = 0; e < 8; ++e) hv[e] = fmaxf(acc[e] + bv, 0.0f);
    u32x4 packed;
    packed[0] = pk2(hv[0], hv[1]); packed[1] = pk2(hv[2], hv[3]);
    packed[2] = pk2(hv[4], hv[5]); packed[3] = pk2(hv[6], hv[7]);
    *(u32x4*)(ht_out + (size_t)(nt * 16 + lr) * NN + i0 + 8 * lh) = packed;
#pragma unroll
    for (int e = 0; e < 8; ++e)
      hb_out[(size_t)(i0 + e + 8 * lh) * DD + nt * 16 + lr] = (uint16_t)f2bf(hv[e]);
  }
}

// ---------------------------------------------------------------------------
// Softmax over D=128 per row. One wave per row.
__global__ void softmax_kernel(const uint16_t* __restrict__ hb, float* __restrict__ out) {
  int wid  = (blockIdx.x * blockDim.x + threadIdx.x) >> 5;
  int lane = threadIdx.x & 31;
  u32x2 hq = *(const u32x2*)(hb + (size_t)wid * DD + lane * 4);
  float v0 = bf2f(hq[0] & 0xFFFFu), v1 = bf2f(hq[0] >> 16);
  float v2 = bf2f(hq[1] & 0xFFFFu), v3 = bf2f(hq[1] >> 16);
  float m = fmaxf(fmaxf(v0, v1), fmaxf(v2, v3));
#pragma unroll
  for (int s = 16; s >= 1; s >>= 1) m = fmaxf(m, __shfl_xor(m, s, 32));
  float e0 = __expf(v0 - m), e1 = __expf(v1 - m), e2 = __expf(v2 - m), e3 = __expf(v3 - m);
  float ssum = e0 + e1 + e2 + e3;
#pragma unroll
  for (int s = 16; s >= 1; s >>= 1) ssum += __shfl_xor(ssum, s, 32);
  float inv = 1.0f / ssum;
  f32x4 o; o[0] = e0 * inv; o[1] = e1 * inv; o[2] = e2 * inv; o[3] = e3 * inv;
  *(f32x4*)(out + (size_t)wid * DD + lane * 4) = o;
}

// ---------------------------------------------------------------------------
extern "C" void kernel_launch(void* const* d_in, const int* in_sizes, int n_in,
                              void* d_out, int out_size, void* d_ws, size_t ws_size,
                              hipStream_t stream) {
  (void)in_sizes; (void)n_in; (void)out_size; (void)ws_size;
  const int*   adj = (const int*)d_in[0];
  const float* x   = (const float*)d_in[1];
  // d_in[2] = labels (unused by reference output)
  const float* W   = (const float*)d_in[3];
  const float* b   = (const float*)d_in[4];
  float* out = (float*)d_out;

  uint8_t* ws = (uint8_t*)d_ws;
  size_t off = 0;
  auto alloc = [&](size_t bytes) -> void* {
    void* p = ws + off;
    off += (bytes + 255) & ~(size_t)255;
    return p;
  };
  uint16_t* xn       = (uint16_t*)alloc((size_t)NN * DD * 2);   // 2 MB
  uint16_t* htX      = (uint16_t*)alloc((size_t)NN * DD * 2);
  uint16_t* sbuf     = (uint16_t*)alloc((size_t)NN * DD * 2);
  uint16_t* ht1      = (uint16_t*)alloc((size_t)NN * DD * 2);
  uint16_t* hb1      = (uint16_t*)alloc((size_t)NN * DD * 2);
  uint16_t* ht2      = (uint16_t*)alloc((size_t)NN * DD * 2);
  uint16_t* hb2      = (uint16_t*)alloc((size_t)NN * DD * 2);
  uint16_t* Wt       = (uint16_t*)alloc((size_t)DD * DD * 2);
  float*    invdeg   = (float*)   alloc((size_t)NN * 4);
  uint32_t* adjbits  = (uint32_t*)alloc((size_t)NN * NW * 4);   // 8 MB
  uint32_t* maskbits = (uint32_t*)alloc((size_t)NN * NW * 4);   // 8 MB

  dim3 blk(256);
  pack_x_kernel  <<<NN * 32 / 256,                blk, 0, stream>>>(x, xn, htX);
  pack_w_kernel  <<<DD * DD / 256,                blk, 0, stream>>>(W, Wt);
  adj_pack_kernel<<<(int)((size_t)NN * NN / 256), blk, 0, stream>>>(adj, adjbits);
  sim_kernel     <<<512 * 256 * 32 / 256,         blk, 0, stream>>>(xn, adjbits, maskbits);
  deg_kernel     <<<NN / 256,                     blk, 0, stream>>>(maskbits, invdeg);
  // layer 1 (h = x)
  agg_kernel     <<<256,                          blk, 0, stream>>>(maskbits, htX, invdeg, sbuf);
  upd_kernel     <<<256,                          blk, 0, stream>>>(sbuf, Wt, b, hb1, ht1);
  // layer 2
  agg_kernel     <<<256,                          blk, 0, stream>>>(maskbits, ht1, invdeg, sbuf);
  upd_kernel     <<<256,                          blk, 0, stream>>>(sbuf, Wt, b, hb2, ht2);
  softmax_kernel <<<NN * 32 / 256,                blk, 0, stream>>>(hb2, out);
}
